// VectorQuantizer_24988119728218
// MI455X (gfx1250) — compile-verified, matrix operand order checked by reading, not run on hardware
//
#include <hip/hip_runtime.h>
#include <hip/hip_bf16.h>

// VQ-VAE vector quantizer for gfx1250 (MI455X), wave32 + WMMA bf16 +
// async global->LDS staging of the shared B (codebook) tiles.
// z_e: (32,256,32,32) f32, embedding: (1024,256) f32.
// Outputs (concat, f32): z_q_st (8388608) | codes (32768) | commit_loss (1).

typedef __attribute__((ext_vector_type(16))) __bf16 v16bf;
typedef __attribute__((ext_vector_type(8)))  float  v8f;

#define NB 32
#define DD 256
#define HW 1024                 // 32*32
#define NN 32768                // NB*HW
#define KK 1024
#define NTILES (NN / 16)        // 2048 waves
#define CTILES (KK / 16)        // 64 code tiles
#define KCH 8                   // 256 / 32 k-chunks
#define TILE_USH (16 * DD)      // ushorts per (hi or lo) code tile = 8KB

union FragA { v16bf v; unsigned short u[16]; };
union FragB { v16bf v; uint4 q[2]; };

__device__ __forceinline__ unsigned short f32_to_bf16_rne(float x, float* hi_f) {
    unsigned int u = __float_as_uint(x);
    unsigned int r = (u + 0x7FFFu + ((u >> 16) & 1u)) & 0xFFFF0000u;
    if (hi_f) *hi_f = __uint_as_float(r);
    return (unsigned short)(r >> 16);
}

__device__ __forceinline__ void async_wait0() {
    asm volatile("s_wait_asynccnt 0x0" ::: "memory");
}

// Issue 2x16B async global->LDS copies (base and base+4096: INST_OFFSET is
// added to BOTH the global and LDS addresses per the CDNA5 async semantics).
__device__ __forceinline__ void async_copy_2x(unsigned lds_byte,
                                              unsigned long long gaddr) {
    asm volatile("global_load_async_to_lds_b128 %0, %1, off\n\t"
                 "global_load_async_to_lds_b128 %0, %1, off offset:4096"
                 :: "v"(lds_byte), "v"(gaddr) : "memory");
}

// ---------------- Kernel 1: embedding -> bf16 hi/lo + norms ----------------
__global__ void vq_prep(const float* __restrict__ emb,
                        unsigned short* __restrict__ ebh,
                        unsigned short* __restrict__ ebl,
                        float* __restrict__ enorm) {
    int k = blockIdx.x, t = threadIdx.x;
    float e = emb[k * DD + t];
    float fh;
    unsigned short h = f32_to_bf16_rne(e, &fh);
    unsigned short l = f32_to_bf16_rne(e - fh, nullptr);
    ebh[k * DD + t] = h;
    ebl[k * DD + t] = l;
    __shared__ float red[256];
    red[t] = e * e;
    __syncthreads();
    for (int s = 128; s > 0; s >>= 1) {
        if (t < s) red[t] += red[t + s];
        __syncthreads();
    }
    if (t == 0) enorm[k] = red[0];
}

// ---------------- Kernel 2: WMMA distance GEMM + argmin --------------------
// One wave owns a 16-row tile of z_flat vs all 1024 codes; the 8 waves of a
// block share each 16-code B tile via double-buffered async LDS staging.
// score(n,k) = ||e_k||^2 - 2 * z_n . e_k   (||z||^2 constant per row)
__global__ void __launch_bounds__(256)
vq_gemm_argmin(const float* __restrict__ z,
               const unsigned short* __restrict__ ebh,
               const unsigned short* __restrict__ ebl,
               const float* __restrict__ enorm,
               int* __restrict__ codesI,
               float* __restrict__ codesF) {
    // [stage][hi=0/lo=1][code][k]
    __shared__ __align__(16) unsigned short sbuf[2][2][TILE_USH];

    const int t     = threadIdx.x;
    const int lane  = t & 31;
    const int wid   = (blockIdx.x * blockDim.x + t) >> 5;  // 0..2047
    const int n0    = wid * 16;
    const int col   = lane & 15;                 // N index / M-subrow selector
    const int khalf = (lane < 16) ? 0 : 8;       // per-ISA 16-bit A/B lane split

    // A operand: z row = n0+col; element d lives at stride HW in z_e layout.
    const int   row  = n0 + col;
    const int   bidx = row >> 10;
    const int   hw   = row & (HW - 1);
    const float* zrow = z + (size_t)bidx * DD * HW + hw;  // zrow[d*HW]

    // Per-thread staging addresses: thread t copies 16B at t*16 and t*16+4096
    // of each 8KB tile (hi and lo).
    const unsigned ldsH[2] = {
        (unsigned)(uintptr_t)&sbuf[0][0][0] + (unsigned)t * 16u,
        (unsigned)(uintptr_t)&sbuf[1][0][0] + (unsigned)t * 16u };
    const unsigned ldsL[2] = {
        (unsigned)(uintptr_t)&sbuf[0][1][0] + (unsigned)t * 16u,
        (unsigned)(uintptr_t)&sbuf[1][1][0] + (unsigned)t * 16u };

    auto fill = [&](int s, int ct) {
        const unsigned short* srcH = ebh + (size_t)ct * TILE_USH + t * 8;
        const unsigned short* srcL = ebl + (size_t)ct * TILE_USH + t * 8;
        async_copy_2x(ldsH[s], (unsigned long long)srcH);
        async_copy_2x(ldsL[s], (unsigned long long)srcL);
    };

    // Preload & split A fragments for all 8 k-chunks (hoisted, registers).
    FragA ah[KCH], al[KCH];
#pragma unroll
    for (int kc = 0; kc < KCH; ++kc) {
#pragma unroll
        for (int j = 0; j < 16; ++j) {
            // elems 0..7 -> K = khalf+0..7 ; elems 8..15 -> K = khalf+16..23
            int d = kc * 32 + khalf + (j < 8 ? j : j + 8);
            float x = zrow[(size_t)d * HW];
            float fh;
            ah[kc].u[j] = f32_to_bf16_rne(x, &fh);
            al[kc].u[j] = f32_to_bf16_rne(x - fh, nullptr);
        }
    }

    float best[8];
    int   bcode[8];
#pragma unroll
    for (int v = 0; v < 8; ++v) { best[v] = 3.4e38f; bcode[v] = 0; }

    fill(0, 0);
    async_wait0();
    __syncthreads();

    for (int ct = 0; ct < CTILES; ++ct) {
        const int s = ct & 1;
        if (ct + 1 < CTILES) fill(s ^ 1, ct + 1);   // overlap copy with WMMAs

        const int code = ct * 16 + col;             // this lane's B column
        const unsigned short* bh0 = &sbuf[s][0][col * DD];
        const unsigned short* bl0 = &sbuf[s][1][col * DD];
        v8f acc = {};
#pragma unroll
        for (int kc = 0; kc < KCH; ++kc) {
            FragB bh, bl;
            const int off = kc * 32 + khalf;        // 16B-aligned (khalf 0/8)
            bh.q[0] = *(const uint4*)(bh0 + off);
            bh.q[1] = *(const uint4*)(bh0 + off + 16);
            bl.q[0] = *(const uint4*)(bl0 + off);
            bl.q[1] = *(const uint4*)(bl0 + off + 16);
            // bf16x3: zh*eh + zh*el + zl*eh ~ fp32-accurate dot product
            acc = __builtin_amdgcn_wmma_f32_16x16x32_bf16(
                      false, ah[kc].v, false, bh.v, (short)0, acc, false, false);
            acc = __builtin_amdgcn_wmma_f32_16x16x32_bf16(
                      false, ah[kc].v, false, bl.v, (short)0, acc, false, false);
            acc = __builtin_amdgcn_wmma_f32_16x16x32_bf16(
                      false, al[kc].v, false, bh.v, (short)0, acc, false, false);
        }
        const float en = enorm[code];
#pragma unroll
        for (int v = 0; v < 8; ++v) {
            float sc = en - 2.0f * acc[v];
            if (sc < best[v]) { best[v] = sc; bcode[v] = code; }
        }

        async_wait0();       // next stage's copies landed (issued pre-compute)
        __syncthreads();     // visible block-wide before anyone reads them
    }

    // Butterfly min-reduce across the 16 lanes of each half (rows 0-7 / 8-15),
    // ties -> lowest code index (matches jnp.argmin).
#pragma unroll
    for (int v = 0; v < 8; ++v) {
        float sc = best[v];
        int   c  = bcode[v];
#pragma unroll
        for (int off = 8; off >= 1; off >>= 1) {
            float os = __shfl_xor(sc, off, 16);
            int   oc = __shfl_xor(c, off, 16);
            if (os < sc || (os == sc && oc < c)) { sc = os; c = oc; }
        }
        if (col == 0) {
            int m = v + khalf;                   // row within tile
            codesI[n0 + m] = c;
            codesF[n0 + m] = (float)c;
        }
    }
}

// ---------------- Kernel 3: gather + STE out + partial loss ----------------
// One block per (b,d) pair; threads sweep hw -> coalesced z_e read / z_q write,
// embedding gathers served from L2 (1 MB resident).
__global__ void vq_gather(const float* __restrict__ z,
                          const float* __restrict__ emb,
                          const int* __restrict__ codesI,
                          float* __restrict__ zq,
                          float* __restrict__ part) {
    const int p = blockIdx.x;            // 0..8191 : b*256 + d
    const int b = p >> 8;
    const int d = p & 255;
    const int t = threadIdx.x;
    const size_t base = (size_t)p * HW;  // (b*DD+d)*HW
    float s = 0.f;
#pragma unroll
    for (int i = 0; i < 4; ++i) {
        int hw = t + i * 256;
        int code = codesI[b * HW + hw];
        float eq = emb[(size_t)code * DD + d];
        float ze = z[base + hw];
        zq[base + hw] = ze + (eq - ze);  // straight-through (mimic fp32 rounding)
        float df = eq - ze;
        s += df * df;
    }
    __shared__ float red[256];
    red[t] = s;
    __syncthreads();
    for (int k = 128; k > 0; k >>= 1) {
        if (t < k) red[t] += red[t + k];
        __syncthreads();
    }
    if (t == 0) part[p] = red[0];
}

// ---------------- Kernel 4: deterministic final loss -----------------------
__global__ void vq_loss(const float* __restrict__ part, float* __restrict__ out) {
    const int t = threadIdx.x;
    float s = 0.f;
    for (int i = t; i < 8192; i += 256) s += part[i];
    __shared__ float red[256];
    red[t] = s;
    __syncthreads();
    for (int k = 128; k > 0; k >>= 1) {
        if (t < k) red[t] += red[t + k];
        __syncthreads();
    }
    if (t == 0) out[0] = 1.25f * red[0] / 8388608.0f;  // (1+BETA)*mean
}

extern "C" void kernel_launch(void* const* d_in, const int* in_sizes, int n_in,
                              void* d_out, int out_size, void* d_ws, size_t ws_size,
                              hipStream_t stream) {
    const float* z   = (const float*)d_in[0];   // 32*256*32*32
    const float* emb = (const float*)d_in[1];   // 1024*256

    // Workspace layout (needs ~1.22 MB):
    char* ws = (char*)d_ws;
    unsigned short* ebh   = (unsigned short*)(ws);                       // 512 KB
    unsigned short* ebl   = (unsigned short*)(ws + (512u << 10));        // 512 KB
    float*          enorm = (float*)(ws + (1024u << 10));                // 4 KB
    int*            codesI= (int*)  (ws + (1024u << 10) + 4096);         // 128 KB
    float*          part  = (float*)(ws + (1024u << 10) + 4096 + (128u << 10)); // 32 KB

    float* zq     = (float*)d_out;              // 8388608
    float* codesF = zq + 8388608;               // 32768
    float* loss   = codesF + 32768;             // 1

    vq_prep<<<KK, 256, 0, stream>>>(emb, ebh, ebl, enorm);
    vq_gemm_argmin<<<NTILES / 8, 256, 0, stream>>>(z, ebh, ebl, enorm, codesI, codesF);
    vq_gather<<<NB * DD, 256, 0, stream>>>(z, emb, codesI, zq, part);
    vq_loss<<<1, 256, 0, stream>>>(part, loss);
}
